// SwinTransformerBlock_45543833207276
// MI455X (gfx1250) — compile-verified, matrix operand order checked by reading, not run on hardware
//
#include <hip/hip_runtime.h>
#include <hip/hip_bf16.h>
#include <math.h>
#include <cstdint>

// ---------------------------------------------------------------------------
// Swin block for MI455X / gfx1250, wave32 + v_wmma_f32_16x16x32_f16 +
// global_load_async_to_lds_b128 (ASYNCcnt) double-buffered GEMM pipeline.
// Pipeline:
//   0) convert weights f32 -> f16, pre-transposed to [N][K]
//   1) window-partition x -> xwh (f16), xph = xw+pos (f16)
//   2) q  = xph @ Wq^T  (WMMA GEMM, f32 out)
//   3) kv = xwh @ Wkv^T (WMMA GEMM, f32 out)
//   4) per-(window,head) attention in LDS (l2norm, WMMA S, softmax, WMMA O) -> oh f16
//   5) proj = oh @ Wp^T (WMMA GEMM, f32 out)
//   6) x1 = x + LN(window_reverse(proj))  -> f32 + f16 copy
//   7) h  = gelu(x1h @ W1^T) (WMMA GEMM, f16 out, fused exact GELU)
//   8) mlp = h @ W2^T (WMMA GEMM, f32 out)
//   9) out = x1 + LN(mlp)
// ---------------------------------------------------------------------------

typedef _Float16 v16h __attribute__((ext_vector_type(16)));
typedef float    v8f  __attribute__((ext_vector_type(8)));

#define NB    32
#define CC    256
#define ROWS  131072            // 32 * 64 * 64 tokens (window-major rows)
#define HID   1024
#define NHEAD 8
#define DHEAD 32
#define SCALE_C (0.0625f)       // 1/sqrt(256)

// Load a 16-half WMMA fragment from LDS following the CDNA5 16-bit A/B layout:
// this lane's row pointer p (already offset by koff), halfs [0..7] and [16..23].
__device__ __forceinline__ v16h lds_frag(const _Float16* p) {
    union { uint4 u[2]; v16h v; } f;
    f.u[0] = *reinterpret_cast<const uint4*>(p);
    f.u[1] = *reinterpret_cast<const uint4*>(p + 16);
    return f.v;
}

// Async 16-byte global -> LDS copy (per-lane), tracked by ASYNCcnt.
// lds = wave-relative LDS byte address (low 32 bits of generic pointer).
__device__ __forceinline__ void async_b128(uint32_t lds, const void* gp) {
    asm volatile("global_load_async_to_lds_b128 %0, %1, off"
                 :: "v"(lds), "v"(gp) : "memory");
}
__device__ __forceinline__ void wait_async0() {
    asm volatile("s_wait_asynccnt 0x0" ::: "memory");
}
__device__ __forceinline__ uint32_t lds_addr(const void* p) {
    return (uint32_t)(uintptr_t)p;
}

// ---------------------------------------------------------------------------
// Weight convert + transpose: W[K][N] f32 -> Wt[N][K] f16
// ---------------------------------------------------------------------------
__global__ void cvt_wt(const float* __restrict__ W, _Float16* __restrict__ Wt,
                       int K, int N) {
    int i = blockIdx.x * blockDim.x + threadIdx.x;
    if (i >= K * N) return;
    int n = i / K;
    int k = i - n * K;
    Wt[(size_t)n * K + k] = (_Float16)W[(size_t)k * N + n];
}

// ---------------------------------------------------------------------------
// Window partition: x(B, H*W, C) -> xwh(Rows, C) f16 and xph = xw + pos
// ---------------------------------------------------------------------------
__global__ void prep_xw(const float* __restrict__ x, const float* __restrict__ pos,
                        _Float16* __restrict__ xwh, _Float16* __restrict__ xph) {
    const int R = blockIdx.x;           // window-major row
    const int c = threadIdx.x;          // channel
    const int widx = R >> 6, t = R & 63;
    const int b  = widx >> 6;
    const int hb = (widx >> 3) & 7;
    const int wb = widx & 7;
    const int hw = t >> 3, ww = t & 7;
    const int s = (b << 12) + ((hb * 8 + hw) << 6) + (wb * 8 + ww);
    const float xv = x[(size_t)s * CC + c];
    xwh[(size_t)R * CC + c] = (_Float16)xv;
    xph[(size_t)R * CC + c] = (_Float16)(xv + pos[t * CC + c]);
}

// ---------------------------------------------------------------------------
// WMMA GEMM: C[M][N] = act(A[M][K] @ Bt[N][K]^T + bias), A/Bt f16 row-major.
// Block = 128 threads (4 wave32), tile 128x64, K-step 32.
// Double-buffered LDS fed by global_load_async_to_lds_b128 (ASYNCcnt).
// Each wave computes a 32x64 strip: 8 WMMAs per K-step, 6 fragment loads.
// ---------------------------------------------------------------------------
template <int OUT_F16, int ACT_GELU>
__global__ void gemm_wmma(const _Float16* __restrict__ A,
                          const _Float16* __restrict__ Bt,
                          const float* __restrict__ bias,
                          void* __restrict__ Cout,
                          int M, int N, int K) {
    __shared__ __align__(16) _Float16 As[2][128][32];
    __shared__ __align__(16) _Float16 Bs[2][64][32];

    const int tid  = threadIdx.x;
    const int wave = tid >> 5;
    const int lane = tid & 31;
    const int lrow = lane & 15;
    const int koff = (lane < 16) ? 0 : 8;

    const int m_blk = blockIdx.y * 128;
    const int n_blk = blockIdx.x * 64;

    // A loader: thread t fetches its full 32-half row (4 x b128)
    const _Float16* __restrict__ gA = A + (size_t)(m_blk + tid) * K;
    // B loader: row = tid>>1, 16-half segment = (tid&1)*16 (2 x b128)
    const int br = tid >> 1, bc = (tid & 1) * 16;
    const _Float16* __restrict__ gB = Bt + (size_t)(n_blk + br) * K + bc;

    const uint32_t la[2] = { lds_addr(&As[0][tid][0]), lds_addr(&As[1][tid][0]) };
    const uint32_t lb[2] = { lds_addr(&Bs[0][br][bc]), lds_addr(&Bs[1][br][bc]) };

    auto issue_tile = [&](int k0, int b) {
        const _Float16* pa = gA + k0;
        async_b128(la[b],      pa);
        async_b128(la[b] + 16, pa + 8);
        async_b128(la[b] + 32, pa + 16);
        async_b128(la[b] + 48, pa + 24);
        const _Float16* pb = gB + k0;
        async_b128(lb[b],      pb);
        async_b128(lb[b] + 16, pb + 8);
    };

    v8f acc[2][4] = {};

    issue_tile(0, 0);
    wait_async0();
    __syncthreads();

    const int nk = K >> 5;
    for (int t = 0; t < nk; ++t) {
        const int cur = t & 1;
        if (t + 1 < nk) issue_tile((t + 1) << 5, cur ^ 1);

        const v16h af0 = lds_frag(&As[cur][wave * 32 + lrow][koff]);
        const v16h af1 = lds_frag(&As[cur][wave * 32 + 16 + lrow][koff]);
#pragma unroll
        for (int j = 0; j < 4; ++j) {
            const v16h bf = lds_frag(&Bs[cur][j * 16 + lrow][koff]);
            acc[0][j] = __builtin_amdgcn_wmma_f32_16x16x32_f16(
                false, af0, false, bf, (short)0, acc[0][j], false, false);
            acc[1][j] = __builtin_amdgcn_wmma_f32_16x16x32_f16(
                false, af1, false, bf, (short)0, acc[1][j], false, false);
        }
        wait_async0();        // next tile fully landed in LDS
        __syncthreads();      // all waves done reading cur, see next
    }

    // epilogue: C layout -> lane<16: M=i, lane>=16: M=8+i; N = j*16 + (lane&15)
#pragma unroll
    for (int sub = 0; sub < 2; ++sub) {
        const int m_base = m_blk + wave * 32 + sub * 16 + ((lane < 16) ? 0 : 8);
#pragma unroll
        for (int j = 0; j < 4; ++j) {
            const int n = n_blk + j * 16 + lrow;
            const float bv = bias ? bias[n] : 0.0f;
#pragma unroll
            for (int i = 0; i < 8; ++i) {
                float v = acc[sub][j][i] + bv;
                if (ACT_GELU) v = 0.5f * v * (1.0f + erff(v * 0.70710678118f));
                const size_t idx = (size_t)(m_base + i) * N + n;
                if (OUT_F16) reinterpret_cast<_Float16*>(Cout)[idx] = (_Float16)v;
                else         reinterpret_cast<float*>(Cout)[idx]    = v;
            }
        }
    }
}

// ---------------------------------------------------------------------------
// Windowed attention, one workgroup per (window, head). 128 threads.
//   qf rows: [win*64+t][256] (head-major cols), kv rows: [win*64+t][512]
//   oh f16 rows: [win*64+t][256]
// ---------------------------------------------------------------------------
__global__ void attn_win(const float* __restrict__ qf,
                         const float* __restrict__ kvf,
                         _Float16* __restrict__ oh) {
    __shared__ __align__(16) _Float16 qs[64][32];
    __shared__ __align__(16) _Float16 ks[64][32];
    __shared__ __align__(16) _Float16 vt[32][64];   // transposed v
    __shared__ __align__(16) float    S[64][64];
    __shared__ __align__(16) _Float16 P[64][64];

    const int tid = threadIdx.x;
    const int w   = blockIdx.x >> 3;
    const int hd  = blockIdx.x & 7;

    // l2norm(q) and l2norm(k) into f16 LDS (threads 0..63 -> q, 64..127 -> k)
    {
        const int r = tid & 63;
        const bool isK = (tid >= 64);
        const float* src = isK ? (kvf + (size_t)(w * 64 + r) * 512 + hd * DHEAD)
                               : (qf  + (size_t)(w * 64 + r) * CC  + hd * DHEAD);
        float buf[DHEAD];
        float ss = 0.0f;
#pragma unroll
        for (int d = 0; d < DHEAD; ++d) { buf[d] = src[d]; ss += buf[d] * buf[d]; }
        const float inv = 1.0f / fmaxf(sqrtf(ss), 1e-12f);
        _Float16* dst = isK ? &ks[r][0] : &qs[r][0];
#pragma unroll
        for (int d = 0; d < DHEAD; ++d) dst[d] = (_Float16)(buf[d] * inv);
    }
    // v -> vt transposed f16
#pragma unroll
    for (int e = 0; e < 16; ++e) {
        const int idx = tid * 16 + e;
        const int r = idx >> 5, d = idx & 31;
        vt[d][r] = (_Float16)kvf[(size_t)(w * 64 + r) * 512 + 256 + hd * DHEAD + d];
    }
    __syncthreads();

    const int wave = tid >> 5, lane = tid & 31;
    const int lrow = lane & 15;
    const int koff = (lane < 16) ? 0 : 8;
    const int mrow = wave * 16 + lrow;
    const int mb   = wave * 16 + ((lane < 16) ? 0 : 8);

    // S = q_hat @ k_hat^T * SCALE   (4 waves x 4 n-tiles, K=32 -> 1 WMMA each)
    {
        const v16h af = lds_frag(&qs[mrow][koff]);
#pragma unroll
        for (int j = 0; j < 4; ++j) {
            v8f acc = {};
            const v16h bf = lds_frag(&ks[j * 16 + lrow][koff]);
            acc = __builtin_amdgcn_wmma_f32_16x16x32_f16(
                false, af, false, bf, (short)0, acc, false, false);
#pragma unroll
            for (int i = 0; i < 8; ++i)
                S[mb + i][j * 16 + lrow] = acc[i] * SCALE_C;
        }
    }
    __syncthreads();

    // softmax over rows (f32), write P as f16
    if (tid < 64) {
        float mx = -1e30f;
#pragma unroll
        for (int n = 0; n < 64; ++n) mx = fmaxf(mx, S[tid][n]);
        float sum = 0.0f;
#pragma unroll
        for (int n = 0; n < 64; ++n) sum += __expf(S[tid][n] - mx);
        const float inv = 1.0f / sum;
#pragma unroll
        for (int n = 0; n < 64; ++n)
            P[tid][n] = (_Float16)(__expf(S[tid][n] - mx) * inv);
    }
    __syncthreads();

    // O = P @ v : per wave 16 rows, 2 n-tiles, K=64 -> 2 WMMAs per tile
#pragma unroll
    for (int j = 0; j < 2; ++j) {
        v8f acc = {};
#pragma unroll
        for (int kc = 0; kc < 2; ++kc) {
            const v16h af = lds_frag(&P[mrow][kc * 32 + koff]);
            const v16h bf = lds_frag(&vt[j * 16 + lrow][kc * 32 + koff]);
            acc = __builtin_amdgcn_wmma_f32_16x16x32_f16(
                false, af, false, bf, (short)0, acc, false, false);
        }
#pragma unroll
        for (int i = 0; i < 8; ++i) {
            const int row = w * 64 + mb + i;
            const int col = hd * DHEAD + j * 16 + lrow;
            oh[(size_t)row * CC + col] = (_Float16)acc[i];
        }
    }
}

// ---------------------------------------------------------------------------
// out = shortcut + LayerNorm(src_row) ; optional window-reverse row mapping.
// One 256-thread block per row.
// ---------------------------------------------------------------------------
__global__ void resln(const float* __restrict__ src, const float* __restrict__ shortcut,
                      const float* __restrict__ g, const float* __restrict__ be,
                      float* __restrict__ out_f32, _Float16* __restrict__ out_f16,
                      int windowed) {
    __shared__ float red[256];
    __shared__ float s_mean, s_rstd;
    const int r = blockIdx.x;
    const int c = threadIdx.x;
    int sr = r;
    if (windowed) {
        const int b = r >> 12, p = r & 4095;
        const int ih = p >> 6, iw = p & 63;
        sr = (((b << 6) + ((ih >> 3) << 3) + (iw >> 3)) << 6) +
             ((ih & 7) << 3) + (iw & 7);
    }
    const float v = src[(size_t)sr * CC + c];

    red[c] = v;
    __syncthreads();
    for (int s = 128; s > 0; s >>= 1) {
        if (c < s) red[c] += red[c + s];
        __syncthreads();
    }
    if (c == 0) s_mean = red[0] * (1.0f / 256.0f);
    __syncthreads();
    const float d = v - s_mean;
    red[c] = d * d;
    __syncthreads();
    for (int s = 128; s > 0; s >>= 1) {
        if (c < s) red[c] += red[c + s];
        __syncthreads();
    }
    if (c == 0) s_rstd = rsqrtf(red[0] * (1.0f / 256.0f) + 1e-5f);
    __syncthreads();

    const float y = d * s_rstd * g[c] + be[c];
    const float o = shortcut[(size_t)r * CC + c] + y;
    out_f32[(size_t)r * CC + c] = o;
    if (out_f16) out_f16[(size_t)r * CC + c] = (_Float16)o;
}

// ---------------------------------------------------------------------------
extern "C" void kernel_launch(void* const* d_in, const int* in_sizes, int n_in,
                              void* d_out, int out_size, void* d_ws, size_t ws_size,
                              hipStream_t stream) {
    const float* x   = (const float*)d_in[0];
    const float* pos = (const float*)d_in[1];
    const float* g1  = (const float*)d_in[2];
    const float* be1 = (const float*)d_in[3];
    const float* g2  = (const float*)d_in[4];
    const float* be2 = (const float*)d_in[5];
    const float* Wq  = (const float*)d_in[6];
    const float* bq  = (const float*)d_in[7];
    const float* Wkv = (const float*)d_in[8];
    const float* bkv = (const float*)d_in[9];
    const float* Wp  = (const float*)d_in[10];
    const float* bp  = (const float*)d_in[11];
    const float* W1  = (const float*)d_in[12];
    const float* b1  = (const float*)d_in[13];
    const float* W2  = (const float*)d_in[14];
    const float* b2  = (const float*)d_in[15];

    // ---- workspace carve (with lifetime-based reuse) ----
    char* p = (char*)d_ws;
    size_t off = 0;
    auto carve = [&](size_t bytes) {
        void* q = p + off;
        off += (bytes + 255) & ~(size_t)255;
        return q;
    };
    _Float16* Wqt  = (_Float16*)carve((size_t)CC * CC * 2);
    _Float16* Wkvt = (_Float16*)carve((size_t)2 * CC * CC * 2);
    _Float16* Wpt  = (_Float16*)carve((size_t)CC * CC * 2);
    _Float16* W1t  = (_Float16*)carve((size_t)HID * CC * 2);
    _Float16* W2t  = (_Float16*)carve((size_t)CC * HID * 2);

    void* bufA  = carve((size_t)ROWS * CC * 2);     //  64MB: xwh -> oh
    void* bufB  = carve((size_t)ROWS * CC * 2);     //  64MB: xph -> x1h
    void* bufQ  = carve((size_t)ROWS * CC * 4);     // 128MB: q -> proj -> mlp
    void* bufKV = carve((size_t)ROWS * 2 * CC * 4); // 256MB: kv -> hidden(f16)
    float* x1   = (float*)carve((size_t)ROWS * CC * 4); // 128MB

    _Float16* xwh = (_Float16*)bufA;
    _Float16* xph = (_Float16*)bufB;
    float*    qf  = (float*)bufQ;
    float*    kvf = (float*)bufKV;
    _Float16* ohh = (_Float16*)bufA;   // reuse (xwh dead after kv GEMM)
    float*    pj  = (float*)bufQ;      // reuse (q dead after attention)
    _Float16* x1h = (_Float16*)bufB;   // reuse (xph dead after q GEMM)
    _Float16* hh  = (_Float16*)bufKV;  // reuse (kv dead after attention)
    float*    mlp = (float*)bufQ;      // reuse (proj dead after resln#1)

    // 0) weights -> f16 transposed
    cvt_wt<<<(CC * CC + 255) / 256, 256, 0, stream>>>(Wq,  Wqt,  CC, CC);
    cvt_wt<<<(CC * 2 * CC + 255) / 256, 256, 0, stream>>>(Wkv, Wkvt, CC, 2 * CC);
    cvt_wt<<<(CC * CC + 255) / 256, 256, 0, stream>>>(Wp,  Wpt,  CC, CC);
    cvt_wt<<<(CC * HID + 255) / 256, 256, 0, stream>>>(W1,  W1t,  CC, HID);
    cvt_wt<<<(HID * CC + 255) / 256, 256, 0, stream>>>(W2,  W2t,  HID, CC);

    // 1) window partition + pos
    prep_xw<<<ROWS, 256, 0, stream>>>(x, pos, xwh, xph);

    // 2) q = xph @ Wq^T + bq           (131072 x 256 x 256)
    gemm_wmma<0, 0><<<dim3(CC / 64, ROWS / 128), 128, 0, stream>>>(
        xph, Wqt, bq, qf, ROWS, CC, CC);
    // 3) kv = xwh @ Wkv^T + bkv        (131072 x 512 x 256)
    gemm_wmma<0, 0><<<dim3(2 * CC / 64, ROWS / 128), 128, 0, stream>>>(
        xwh, Wkvt, bkv, kvf, ROWS, 2 * CC, CC);

    // 4) attention -> oh (f16)
    attn_win<<<2048 * NHEAD, 128, 0, stream>>>(qf, kvf, ohh);

    // 5) proj = oh @ Wp^T + bp         (131072 x 256 x 256)
    gemm_wmma<0, 0><<<dim3(CC / 64, ROWS / 128), 128, 0, stream>>>(
        ohh, Wpt, bp, pj, ROWS, CC, CC);

    // 6) x1 = x + LN(window_reverse(proj))
    resln<<<ROWS, 256, 0, stream>>>(pj, x, g1, be1, x1, x1h, 1);

    // 7) h = gelu(x1 @ W1^T + b1)      (131072 x 1024 x 256), f16 out
    gemm_wmma<1, 1><<<dim3(HID / 64, ROWS / 128), 128, 0, stream>>>(
        x1h, W1t, b1, hh, ROWS, HID, CC);
    // 8) mlp = h @ W2^T + b2           (131072 x 256 x 1024)
    gemm_wmma<0, 0><<<dim3(CC / 64, ROWS / 128), 128, 0, stream>>>(
        hh, W2t, b2, mlp, ROWS, CC, HID);

    // 9) out = x1 + LN(mlp)
    resln<<<ROWS, 256, 0, stream>>>(mlp, x1, g2, be2, (float*)d_out, nullptr, 0);
}
